// GraphSAGE_5574867550247
// MI455X (gfx1250) — matrix operand order repaired and checked by reading, not compiled, not run on hardware
//
#include <hip/hip_runtime.h>
#include <hip/hip_bf16.h>

// ---------------------------------------------------------------------------
// GraphSAGE forward for MI455X (gfx1250, wave32).
//
// Cost model: dense math is ~3.7 GFLOP (trivial for WMMA); the edge scatter
// (3 x 800k edges x 64 f32 gather + f32 atomic-add) dominates and the whole
// ~40MB working set lives in the 192MB L2, so we are L2-bandwidth bound.
// Strategy: coalesced float4 scatter, then ONE fused WMMA kernel per layer
// that does  t = mean@Wl + x@Wr + bl ;  h = relu(t@Wm + bm)  and the graph
// pooling accumulation in a single pass (24 v_wmma_f32_16x16x32_f16 / wave).
// ---------------------------------------------------------------------------

#define N_NODES  50000
#define N_EDGES  800000
#define D        64
#define N_GRAPHS 512
#define D_CAT    192
#define D_TARGET 10

typedef __attribute__((ext_vector_type(16))) _Float16 v16h;
typedef __attribute__((ext_vector_type(8)))  _Float16 v8h;
typedef __attribute__((ext_vector_type(8)))  float    v8f;

// ----------------------------- utility kernels -----------------------------

__global__ void zero_f32(float* __restrict__ p, int n) {
    int i = blockIdx.x * blockDim.x + threadIdx.x;
    if (i < n) p[i] = 0.0f;
}

__global__ void deg_kernel(const int* __restrict__ dst, float* __restrict__ deg) {
    int e = blockIdx.x * blockDim.x + threadIdx.x;
    if (e < N_EDGES) atomicAdd(&deg[dst[e]], 1.0f);
}

__global__ void deginv_kernel(float* __restrict__ deg) {
    int v = blockIdx.x * blockDim.x + threadIdx.x;
    if (v < N_NODES) deg[v] = 1.0f / fmaxf(deg[v], 1.0f);
}

__global__ void gcnt_kernel(const int* __restrict__ batch, float* __restrict__ gcnt) {
    int v = blockIdx.x * blockDim.x + threadIdx.x;
    if (v < N_NODES) atomicAdd(&gcnt[batch[v]], 1.0f);
}

// Pre-pack a 64x64 f32 weight matrix into WMMA B-fragment lane layout (f16).
// B frag (16x16x32 f16): lane = halfsel*16 + n ; element i <-> W[kf*32+8*halfsel+i+(i>=8?8:0), n]
// Stored as [kf*4+ntile][lane][16 halfs] -> fragment load = contiguous 32B.
__global__ void pack_w(const float* __restrict__ W, _Float16* __restrict__ out) {
    int lane = threadIdx.x;                 // 32 lanes
    int nt = blockIdx.x & 3, kf = blockIdx.x >> 2;   // grid = 8
    int n  = nt * 16 + (lane & 15);
    int hf = lane >> 4;
    _Float16* dstp = out + ((kf * 4 + nt) * 32 + lane) * 16;
#pragma unroll
    for (int i = 0; i < 16; ++i)
        dstp[i] = (_Float16)W[(kf * 32 + 8 * hf + i + (i >= 8 ? 8 : 0)) * D + n];
}

// Edge scatter: 16 threads per edge, each moves one float4 of the source row.
__global__ void scatter_kernel(const float* __restrict__ hin,
                               const int* __restrict__ src,
                               const int* __restrict__ dstn,
                               float* __restrict__ agg) {
    int t = blockIdx.x * blockDim.x + threadIdx.x;
    int e = t >> 4;
    if (e >= N_EDGES) return;
    int f = (t & 15) * 4;
    int s = src[e], d = dstn[e];
    const float4 v = *(const float4*)(hin + (size_t)s * D + f);
    float* o = agg + (size_t)d * D + f;
    atomicAdd(o + 0, v.x);
    atomicAdd(o + 1, v.y);
    atomicAdd(o + 2, v.z);
    atomicAdd(o + 3, v.w);
}

// ----------------------------- fused layer kernel --------------------------

__device__ __forceinline__ void cvt8(const float* __restrict__ p, float s,
                                     _Float16* __restrict__ o) {
    float4 a = *(const float4*)p;
    float4 b = *(const float4*)(p + 4);
    o[0] = (_Float16)(a.x * s); o[1] = (_Float16)(a.y * s);
    o[2] = (_Float16)(a.z * s); o[3] = (_Float16)(a.w * s);
    o[4] = (_Float16)(b.x * s); o[5] = (_Float16)(b.y * s);
    o[6] = (_Float16)(b.z * s); o[7] = (_Float16)(b.w * s);
}

// 625 blocks x 160 threads (5 waves); each wave owns one exact 16-row tile
// (50000 = 16 * 3125 = 16 * 625 * 5 -> no tails, uniform barriers).
__global__ __launch_bounds__(160) void sage_layer(
    const float* __restrict__ hin, const float* __restrict__ agg,
    const float* __restrict__ deginv,
    const _Float16* __restrict__ pWl, const _Float16* __restrict__ pWr,
    const _Float16* __restrict__ pWm,
    const float* __restrict__ bl, const float* __restrict__ bm,
    const int* __restrict__ batch,
    float* __restrict__ hout, float* __restrict__ gsum, int layer_off) {
    __shared__ __align__(16) _Float16 ldsT[5][16 * D];   // per-wave 16x64 f16 tile

    const int wv   = threadIdx.x >> 5;
    const int lane = threadIdx.x & 31;
    const int m    = lane & 15;       // A-row / B-col within tile
    const int hf   = lane >> 4;       // K-half selector
    const int tile = blockIdx.x * 5 + wv;
    const int row  = tile * 16 + m;

    const float dinv = deginv[row];

    // --- A fragments for mean (= agg * 1/deg) and x, ISA 16-bit A layout ---
    v16h aM[2], aX[2];
#pragma unroll
    for (int kf = 0; kf < 2; ++kf) {
        _Float16 tmp[16];
        const int c0 = kf * 32 + 8 * hf;
        cvt8(agg + (size_t)row * D + c0,      dinv, tmp);
        cvt8(agg + (size_t)row * D + c0 + 16, dinv, tmp + 8);
#pragma unroll
        for (int i = 0; i < 16; ++i) aM[kf][i] = tmp[i];
        cvt8(hin + (size_t)row * D + c0,      1.0f, tmp);
        cvt8(hin + (size_t)row * D + c0 + 16, 1.0f, tmp + 8);
#pragma unroll
        for (int i = 0; i < 16; ++i) aX[kf][i] = tmp[i];
    }

    // --- GEMM 1: t = mean@Wl + x@Wr  (16 WMMAs, shared f32 accumulators) ---
    v8f c[4] = {};
#pragma unroll
    for (int kf = 0; kf < 2; ++kf) {
#pragma unroll
        for (int nt = 0; nt < 4; ++nt) {
            v16h bL = *(const v16h*)(pWl + ((kf * 4 + nt) * 32 + lane) * 16);
            v16h bR = *(const v16h*)(pWr + ((kf * 4 + nt) * 32 + lane) * 16);
            c[nt] = __builtin_amdgcn_wmma_f32_16x16x32_f16(
                false, aM[kf], false, bL, (short)0, c[nt], false, false);
            c[nt] = __builtin_amdgcn_wmma_f32_16x16x32_f16(
                false, aX[kf], false, bR, (short)0, c[nt], false, false);
        }
    }

    // --- + bl, stage t in LDS (f16) for the C->A lane transpose ---
#pragma unroll
    for (int nt = 0; nt < 4; ++nt) {
        const int col = nt * 16 + m;
        const float blv = bl[col];
#pragma unroll
        for (int r = 0; r < 8; ++r)
            ldsT[wv][(8 * hf + r) * D + col] = (_Float16)(c[nt][r] + blv);
    }
    __syncthreads();

    // --- rebuild A fragments of t from LDS (aligned 16B reads) ---
    v16h aT[2];
    const _Float16* lrow = &ldsT[wv][m * D];
#pragma unroll
    for (int kf = 0; kf < 2; ++kf) {
        v8h lo = *(const v8h*)(lrow + kf * 32 + 8 * hf);
        v8h hi = *(const v8h*)(lrow + kf * 32 + 16 + 8 * hf);
#pragma unroll
        for (int i = 0; i < 8; ++i) { aT[kf][i] = lo[i]; aT[kf][8 + i] = hi[i]; }
    }

    // --- GEMM 2: h = relu(t@Wm + bm)  (8 WMMAs) ---
    v8f c2[4] = {};
#pragma unroll
    for (int kf = 0; kf < 2; ++kf) {
#pragma unroll
        for (int nt = 0; nt < 4; ++nt) {
            v16h bM = *(const v16h*)(pWm + ((kf * 4 + nt) * 32 + lane) * 16);
            c2[nt] = __builtin_amdgcn_wmma_f32_16x16x32_f16(
                false, aT[kf], false, bM, (short)0, c2[nt], false, false);
        }
    }

    int bidx[8];
#pragma unroll
    for (int r = 0; r < 8; ++r) bidx[r] = batch[tile * 16 + 8 * hf + r];

    // --- epilogue: bias + relu + store h + fused graph-pool accumulation ---
#pragma unroll
    for (int nt = 0; nt < 4; ++nt) {
        const int col = nt * 16 + m;
        const float bmv = bm[col];
#pragma unroll
        for (int r = 0; r < 8; ++r) {
            const int rw = tile * 16 + 8 * hf + r;
            float v = fmaxf(c2[nt][r] + bmv, 0.0f);
            hout[(size_t)rw * D + col] = v;
            atomicAdd(&gsum[bidx[r] * D_CAT + layer_off + col], v);
        }
    }
}

// ----------------------------- final MLP (tiny) -----------------------------

__global__ __launch_bounds__(64) void mlp_kernel(
    const float* __restrict__ gsum, const float* __restrict__ gcnt,
    const float* __restrict__ W1, const float* __restrict__ b1,
    const float* __restrict__ W2, const float* __restrict__ b2,
    float* __restrict__ out) {
    __shared__ float gv[D_CAT];
    __shared__ float g1[D];
    const int g = blockIdx.x;
    const int j = threadIdx.x;
    const float inv = 1.0f / fmaxf(gcnt[g], 1.0f);
    for (int k = j; k < D_CAT; k += D) gv[k] = gsum[g * D_CAT + k] * inv;
    __syncthreads();
    float acc = b1[j];
#pragma unroll 4
    for (int k = 0; k < D_CAT; ++k) acc += gv[k] * W1[k * D + j];
    g1[j] = fmaxf(acc, 0.0f);
    __syncthreads();
    if (j < D_TARGET) {
        float o = b2[j];
#pragma unroll 4
        for (int k = 0; k < D; ++k) o += g1[k] * W2[k * D_TARGET + j];
        out[g * D_TARGET + j] = o;
    }
}

// ----------------------------- launch -----------------------------

extern "C" void kernel_launch(void* const* d_in, const int* in_sizes, int n_in,
                              void* d_out, int out_size, void* d_ws, size_t ws_size,
                              hipStream_t stream) {
    (void)in_sizes; (void)n_in; (void)out_size; (void)ws_size;

    const float* x     = (const float*)d_in[0];
    const int*   edge  = (const int*)d_in[1];
    const int*   src   = edge;
    const int*   dst   = edge + N_EDGES;
    const int*   batch = (const int*)d_in[2];
    const float* Wl[3] = {(const float*)d_in[3], (const float*)d_in[6], (const float*)d_in[9]};
    const float* bl[3] = {(const float*)d_in[4], (const float*)d_in[7], (const float*)d_in[10]};
    const float* Wr[3] = {(const float*)d_in[5], (const float*)d_in[8], (const float*)d_in[11]};
    const float* Wm = (const float*)d_in[12];
    const float* bm = (const float*)d_in[13];
    const float* W1 = (const float*)d_in[14];
    const float* b1 = (const float*)d_in[15];
    const float* W2 = (const float*)d_in[16];
    const float* b2 = (const float*)d_in[17];

    // workspace carve-out (all offsets 256B-aligned)
    char* ws = (char*)d_ws;
    float* agg    = (float*)ws; ws += (size_t)N_NODES * D * 4;     // 12.8 MB
    float* hA     = (float*)ws; ws += (size_t)N_NODES * D * 4;     // 12.8 MB
    float* hB     = (float*)ws; ws += (size_t)N_NODES * D * 4;     // 12.8 MB
    float* deginv = (float*)ws; ws += (size_t)50176 * 4;
    float* gsum   = (float*)ws; ws += (size_t)N_GRAPHS * D_CAT * 4;
    float* gcnt   = (float*)ws; ws += (size_t)1024 * 4;
    _Float16* pw  = (_Float16*)ws;                                 // 7 x 8KB packed weights
    _Float16* pWl[3] = {pw,          pw + 2 * 4096, pw + 4 * 4096};
    _Float16* pWr[3] = {pw + 4096,   pw + 3 * 4096, pw + 5 * 4096};
    _Float16* pWm    =  pw + 6 * 4096;

    // degree / graph-count / weight packing (once per launch, deterministic)
    zero_f32<<<(N_NODES + 255) / 256, 256, 0, stream>>>(deginv, N_NODES);
    zero_f32<<<(N_GRAPHS * D_CAT + 255) / 256, 256, 0, stream>>>(gsum, N_GRAPHS * D_CAT);
    zero_f32<<<(N_GRAPHS + 255) / 256, 256, 0, stream>>>(gcnt, N_GRAPHS);
    deg_kernel<<<(N_EDGES + 255) / 256, 256, 0, stream>>>(dst, deginv);
    deginv_kernel<<<(N_NODES + 255) / 256, 256, 0, stream>>>(deginv);
    gcnt_kernel<<<(N_NODES + 255) / 256, 256, 0, stream>>>(batch, gcnt);
    for (int i = 0; i < 3; ++i) {
        pack_w<<<8, 32, 0, stream>>>(Wl[i], pWl[i]);
        pack_w<<<8, 32, 0, stream>>>(Wr[i], pWr[i]);
    }
    pack_w<<<8, 32, 0, stream>>>(Wm, pWm);

    // three fused SAGE layers
    const float* hin = x;
    float* houts[3] = {hA, hB, hA};
    for (int layer = 0; layer < 3; ++layer) {
        zero_f32<<<(N_NODES * D + 255) / 256, 256, 0, stream>>>(agg, N_NODES * D);
        scatter_kernel<<<(N_EDGES * 16 + 255) / 256, 256, 0, stream>>>(hin, src, dst, agg);
        sage_layer<<<625, 160, 0, stream>>>(hin, agg, deginv,
                                            pWl[layer], pWr[layer], pWm,
                                            bl[layer], bm, batch,
                                            houts[layer], gsum, layer * D);
        hin = houts[layer];
    }

    // readout MLP
    mlp_kernel<<<N_GRAPHS, 64, 0, stream>>>(gsum, gcnt, W1, b1, W2, b2, (float*)d_out);
}
